// HTGPModel_27943057228072
// MI455X (gfx1250) — matrix operand order repaired and checked by reading, compiled, not run on hardware
//
#include <hip/hip_runtime.h>
#include <hip/hip_bf16.h>
#include <math.h>

// ---------------------------------------------------------------------------
// PaiNN-like GNN fused for MI455X (gfx1250, wave32, WMMA bf16).
// Edge-side GEMMs (~86 GFLOP) dominate -> v_wmma_f32_16x16x32_bf16 with f32
// accumulate.  64-edge tiles with 4-way M register blocking: each B weight
// fragment is loaded once (two 16B global loads, L2-hot) and reused by 4
// WMMAs, so the kernel is matrix-pipe bound rather than load-pipe bound.
// ---------------------------------------------------------------------------

typedef __bf16 bf16_t;
typedef __attribute__((ext_vector_type(16))) __bf16 v16bf;
typedef __attribute__((ext_vector_type(8)))  __bf16 v8bf;
typedef __attribute__((ext_vector_type(8)))  float  v8f;

#define NN   10000
#define EE   320000
#define GG   16
#define HH   128
#define RR   32
#define LL   3
#define GIN  288            // 2H + R
#define TE   64             // edges per block (4 M-subtiles of 16)
#define CUT  5.0f
#define PI_F 3.14159265358979f

__device__ __forceinline__ float fsilu(float x) { return x / (1.0f + __expf(-x)); }
__device__ __forceinline__ float fsigm(float x) { return 1.0f / (1.0f + __expf(-x)); }

// A fragment (16x32 bf16, M x K) from LDS, row-major [16][kstride].
// ISA 7.12.2 16-bit A layout: lanes 0-15 hold K = k0+{0..7, 16..23},
// lanes 16-31 hold K = k0+8+{0..7, 16..23}.  Two 16-byte LDS loads.
__device__ __forceinline__ v16bf frag_a_lds(const bf16_t* base, int kstride,
                                            int k0, int lane) {
  int m    = lane & 15;
  int kadd = (lane & 16) ? 8 : 0;
  const bf16_t* p = base + m * kstride + k0 + kadd;
  v8bf lo = *(const v8bf*)(p);
  v8bf hi = *(const v8bf*)(p + 16);
  return __builtin_shufflevector(lo, hi, 0, 1, 2, 3, 4, 5, 6, 7,
                                 8, 9, 10, 11, 12, 13, 14, 15);
}

// B fragment (32x16 bf16, K x N) from global weights pre-transposed to [N][K]:
// lanes 0-15 hold K = k0+{0..7, 16..23} for N = n0+lane; lanes 16-31 hold K = k0+8+{0..7, 16..23}.
// Two 16-byte global loads.
__device__ __forceinline__ v16bf frag_b_glb(const bf16_t* wt, int ktot,
                                            int n0, int k0, int lane) {
  int n    = n0 + (lane & 15);
  int kadd = (lane & 16) ? 8 : 0;
  const bf16_t* p = wt + n * ktot + k0 + kadd;
  v8bf lo = *(const v8bf*)(p);
  v8bf hi = *(const v8bf*)(p + 16);
  return __builtin_shufflevector(lo, hi, 0, 1, 2, 3, 4, 5, 6, 7,
                                 8, 9, 10, 11, 12, 13, 14, 15);
}

__device__ __forceinline__ v8f wmma_bf16(v16bf a, v16bf b, v8f c) {
  return __builtin_amdgcn_wmma_f32_16x16x32_bf16(false, a, false, b,
                                                 (short)0, c, false, false);
}

// ---------------------------------------------------------------------------
// Fused edge kernel: one block = 64 edges, 8 waves; wave w owns columns
// [16w, 16w+16) and 4 M-subtiles.  88 WMMAs per wave per block.
// ---------------------------------------------------------------------------
__global__ void __launch_bounds__(256)
k_edge(const int* __restrict__ ei, const float* __restrict__ pos,
       const float* __restrict__ h0, const float* __restrict__ h1,
       const bf16_t* __restrict__ WmsgT, const float* __restrict__ bmsg,
       const bf16_t* __restrict__ WrbfT,
       const bf16_t* __restrict__ Wg1T, const float* __restrict__ bg1,
       const bf16_t* __restrict__ WgvT,
       const bf16_t* __restrict__ Wg2T, const float* __restrict__ bg2,
       float* __restrict__ agg, float* __restrict__ dh1) {
  __shared__ __align__(16) bf16_t sGin[TE * GIN];   // [h0row | h0col | rbf]
  __shared__ __align__(16) bf16_t sBuf[TE * HH];    // s1, then re-used as pre
  __shared__ float sRhat[TE][3];
  __shared__ float sD[TE];
  __shared__ int   sRow[TE], sCol[TE];

  const int tid  = threadIdx.x;
  const int lane = tid & 31;
  const int wave = tid >> 5;
  const int e0   = blockIdx.x * TE;

  if (tid < TE) {
    int r = ei[e0 + tid];
    int c = ei[EE + e0 + tid];
    sRow[tid] = r;
    sCol[tid] = c;
    float dx = pos[c * 3 + 0] - pos[r * 3 + 0];
    float dy = pos[c * 3 + 1] - pos[r * 3 + 1];
    float dz = pos[c * 3 + 2] - pos[r * 3 + 2];
    float d  = sqrtf(dx * dx + dy * dy + dz * dz);
    d = fmaxf(d, 1e-8f);
    sD[tid] = d;
    sRhat[tid][0] = dx / d;
    sRhat[tid][1] = dy / d;
    sRhat[tid][2] = dz / d;
  }
  if (lane == 0) {  // pull this wave's weight slices toward L2/L0
    __builtin_prefetch(WmsgT + (wave * 16) * HH, 0, 1);
    __builtin_prefetch(Wg1T + (wave * 16) * GIN, 0, 1);
  }
  __syncthreads();

  // Gather endpoint scalars (fp32 -> bf16) into g_in.  TE*128 = 8192 elems.
#pragma unroll 4
  for (int j = 0; j < 32; ++j) {
    int idx = tid + 256 * j;
    int e = idx >> 7, c = idx & 127;
    sGin[e * GIN + c]      = (bf16_t)h0[sRow[e] * HH + c];
    sGin[e * GIN + HH + c] = (bf16_t)h0[sCol[e] * HH + c];
  }
  // RBF with cosine cutoff.  TE*32 = 2048 elems.
#pragma unroll 4
  for (int j = 0; j < 8; ++j) {
    int idx = tid + 256 * j;
    int e = idx >> 5, r = idx & 31;
    float d      = sD[e];
    float center = (CUT / (RR - 1)) * (float)r;
    float gamma  = (RR / CUT) * (RR / CUT);
    float fcut   = (d < CUT) ? 0.5f * (__cosf(PI_F * d / CUT) + 1.0f) : 0.0f;
    float t      = d - center;
    sGin[e * GIN + 2 * HH + r] = (bf16_t)(__expf(-gamma * t * t) * fcut);
  }
  // s1[e][h] = sum_c h1[row[e]][c][h] * rhat[e][c]   -> sBuf
#pragma unroll 4
  for (int j = 0; j < 32; ++j) {
    int idx = tid + 256 * j;
    int e = idx >> 7, hcol = idx & 127;
    const float* hp = h1 + sRow[e] * 3 * HH + hcol;
    float v = hp[0] * sRhat[e][0] + hp[HH] * sRhat[e][1] + hp[2 * HH] * sRhat[e][2];
    sBuf[e * HH + hcol] = (bf16_t)v;
  }
  __syncthreads();

  const int n0    = wave * 16;
  const int ncol  = n0 + (lane & 15);
  const int mbase = (lane & 16) ? 8 : 0;

  // ---- m_lin = h0[col] @ W_msg ; rbf @ W_rbf -------------------------------
  v8f accM[4] = {{}, {}, {}, {}};
  v8f accR[4] = {{}, {}, {}, {}};
#pragma unroll
  for (int k = 0; k < HH; k += 32) {
    v16bf b = frag_b_glb(WmsgT, HH, n0, k, lane);
#pragma unroll
    for (int mt = 0; mt < 4; ++mt)
      accM[mt] = wmma_bf16(frag_a_lds(sGin + mt * 16 * GIN, GIN, HH + k, lane),
                           b, accM[mt]);
  }
  {
    v16bf b = frag_b_glb(WrbfT, RR, n0, 0, lane);
#pragma unroll
    for (int mt = 0; mt < 4; ++mt)
      accR[mt] = wmma_bf16(frag_a_lds(sGin + mt * 16 * GIN, GIN, 2 * HH, lane),
                           b, accR[mt]);
  }
  const float bm = bmsg[ncol];
  float mval[4][8];
#pragma unroll
  for (int mt = 0; mt < 4; ++mt)
#pragma unroll
    for (int r = 0; r < 8; ++r)
      mval[mt][r] = (accM[mt][r] + bm) * accR[mt][r];

  // ---- pre = silu(g_in @ Wg1 + bg1 + s1 @ Wgv) -----------------------------
  v8f accG[4] = {{}, {}, {}, {}};
#pragma unroll
  for (int k = 0; k < GIN; k += 32) {
    v16bf b = frag_b_glb(Wg1T, GIN, n0, k, lane);
#pragma unroll
    for (int mt = 0; mt < 4; ++mt)
      accG[mt] = wmma_bf16(frag_a_lds(sGin + mt * 16 * GIN, GIN, k, lane),
                           b, accG[mt]);
  }
  v8f accS[4] = {{}, {}, {}, {}};
#pragma unroll
  for (int k = 0; k < HH; k += 32) {
    v16bf b = frag_b_glb(WgvT, HH, n0, k, lane);
#pragma unroll
    for (int mt = 0; mt < 4; ++mt)
      accS[mt] = wmma_bf16(frag_a_lds(sBuf + mt * 16 * HH, HH, k, lane),
                           b, accS[mt]);
  }
  __syncthreads();  // all waves done reading s1 from sBuf

  const float b1 = bg1[ncol];
#pragma unroll
  for (int mt = 0; mt < 4; ++mt)
#pragma unroll
    for (int r = 0; r < 8; ++r)
      sBuf[(mt * 16 + mbase + r) * HH + ncol] =
          (bf16_t)fsilu(accG[mt][r] + b1 + accS[mt][r]);
  __syncthreads();  // pre visible to all waves

  // ---- gate = sigmoid(pre @ Wg2 + bg2) ; gm = gate * m ; scatter -----------
  v8f acc2[4] = {{}, {}, {}, {}};
#pragma unroll
  for (int k = 0; k < HH; k += 32) {
    v16bf b = frag_b_glb(Wg2T, HH, n0, k, lane);
#pragma unroll
    for (int mt = 0; mt < 4; ++mt)
      acc2[mt] = wmma_bf16(frag_a_lds(sBuf + mt * 16 * HH, HH, k, lane),
                           b, acc2[mt]);
  }
  const float b2 = bg2[ncol];
#pragma unroll
  for (int mt = 0; mt < 4; ++mt) {
#pragma unroll
    for (int r = 0; r < 8; ++r) {
      int   e    = mt * 16 + mbase + r;
      float gm   = fsigm(acc2[mt][r] + b2) * mval[mt][r];
      int   node = sRow[e];
      atomicAdd(&agg[node * HH + ncol], gm);
      atomicAdd(&dh1[node * 3 * HH + 0 * HH + ncol], sRhat[e][0] * gm);
      atomicAdd(&dh1[node * 3 * HH + 1 * HH + ncol], sRhat[e][1] * gm);
      atomicAdd(&dh1[node * 3 * HH + 2 * HH + ncol], sRhat[e][2] * gm);
    }
  }
}

// ---------------------------------------------------------------------------
// Node update + readout: one block = 16 nodes, 8 waves, 12 WMMAs per wave.
// ---------------------------------------------------------------------------
__global__ void __launch_bounds__(256)
k_node(const float* __restrict__ agg, const float* __restrict__ dh1,
       const bf16_t* __restrict__ Wd1T, const float* __restrict__ bd1,
       const bf16_t* __restrict__ Wd2T, const float* __restrict__ bd2,
       const bf16_t* __restrict__ Wr1T, const float* __restrict__ br1,
       const float* __restrict__ Wr2, const float* __restrict__ br2,
       const int* __restrict__ batch,
       float* __restrict__ h0, float* __restrict__ h1,
       float* __restrict__ etot) {
  __shared__ __align__(16) bf16_t sA[16 * HH];
  __shared__ __align__(16) bf16_t sT[16 * HH];
  __shared__ __align__(16) bf16_t sH[16 * HH];
  __shared__ float sE[16];

  const int tid   = threadIdx.x;
  const int lane  = tid & 31;
  const int wave  = tid >> 5;
  const int node0 = blockIdx.x * 16;

  if (tid < 16) sE[tid] = 0.0f;
#pragma unroll
  for (int j = 0; j < 8; ++j) {
    int idx = tid + 256 * j;
    int r = idx >> 7, c = idx & 127;
    sA[r * HH + c] = (bf16_t)agg[(node0 + r) * HH + c];
  }
  // h1 += dh1 (16 nodes x 3 x 128)
#pragma unroll 4
  for (int j = 0; j < 24; ++j) {
    int idx = tid + 256 * j;        // 0..6143
    int off = node0 * 3 * HH + idx;
    h1[off] += dh1[off];
  }
  __syncthreads();

  const int n0    = wave * 16;
  const int ncol  = n0 + (lane & 15);
  const int mbase = (lane & 16) ? 8 : 0;

  v8f a1 = {};
#pragma unroll
  for (int k = 0; k < HH; k += 32)   // agg @ Wd1
    a1 = wmma_bf16(frag_a_lds(sA, HH, k, lane),
                   frag_b_glb(Wd1T, HH, n0, k, lane), a1);
  const float bb1 = bd1[ncol];
#pragma unroll
  for (int r = 0; r < 8; ++r)
    sT[(mbase + r) * HH + ncol] = (bf16_t)fsilu(a1[r] + bb1);
  __syncthreads();

  v8f a2 = {};
#pragma unroll
  for (int k = 0; k < HH; k += 32)   // silu(.) @ Wd2
    a2 = wmma_bf16(frag_a_lds(sT, HH, k, lane),
                   frag_b_glb(Wd2T, HH, n0, k, lane), a2);
  const float bb2 = bd2[ncol];
#pragma unroll
  for (int r = 0; r < 8; ++r) {
    int   node = node0 + mbase + r;
    float hn   = h0[node * HH + ncol] + a2[r] + bb2;
    h0[node * HH + ncol] = hn;
    sH[(mbase + r) * HH + ncol] = (bf16_t)hn;
  }
  __syncthreads();

  v8f a3 = {};
#pragma unroll
  for (int k = 0; k < HH; k += 32)   // h0_new @ Wr1
    a3 = wmma_bf16(frag_a_lds(sH, HH, k, lane),
                   frag_b_glb(Wr1T, HH, n0, k, lane), a3);
  const float bbr = br1[ncol];
  const float w2  = Wr2[ncol];
#pragma unroll
  for (int r = 0; r < 8; ++r)
    atomicAdd(&sE[mbase + r], fsilu(a3[r] + bbr) * w2);  // ds_add_f32
  __syncthreads();

  if (tid < 16) {
    int node = node0 + tid;
    atomicAdd(&etot[batch[node]], sE[tid] + br2[0]);
  }
}

// ---------------------------------------------------------------------------
// Helpers: weight convert/transpose (fp32 [K][N] -> bf16 [N][K]), init, zero.
// ---------------------------------------------------------------------------
__global__ void k_convT(const float* __restrict__ src, bf16_t* __restrict__ dst,
                        int K, int Nc) {
  int idx = blockIdx.x * 256 + threadIdx.x;
  if (idx >= K * Nc) return;
  int n = idx / K, k = idx - n * K;
  dst[n * K + k] = (bf16_t)src[k * Nc + n];
}

__global__ void k_init_h0(const int* __restrict__ z, const float* __restrict__ emb,
                          float* __restrict__ h0) {
  int idx = blockIdx.x * 256 + threadIdx.x;
  if (idx >= NN * HH) return;
  h0[idx] = emb[z[idx >> 7] * HH + (idx & 127)];
}

__global__ void k_zero(float* __restrict__ p, int count) {
  int idx = blockIdx.x * 256 + threadIdx.x;
  if (idx < count) p[idx] = 0.0f;
}

__global__ void k_ref(const int* __restrict__ z, const int* __restrict__ batch,
                      const float* __restrict__ aref, float* __restrict__ etot) {
  int n = blockIdx.x * 256 + threadIdx.x;
  if (n < NN) atomicAdd(&etot[batch[n]], aref[z[n]]);
}

// ---------------------------------------------------------------------------
extern "C" void kernel_launch(void* const* d_in, const int* in_sizes, int n_in,
                              void* d_out, int out_size, void* d_ws, size_t ws_size,
                              hipStream_t stream) {
  (void)in_sizes; (void)n_in; (void)ws_size;
  const int*   z     = (const int*)  d_in[0];
  const float* pos   = (const float*)d_in[1];
  const int*   ei    = (const int*)  d_in[2];
  const int*   batch = (const int*)  d_in[3];
  const float* emb   = (const float*)d_in[4];
  const float* aref  = (const float*)d_in[5];
  const float* W_msg = (const float*)d_in[6];
  const float* b_msg = (const float*)d_in[7];
  const float* W_rbf = (const float*)d_in[8];
  const float* Wg1   = (const float*)d_in[9];
  const float* bg1   = (const float*)d_in[10];
  const float* Wgv   = (const float*)d_in[11];
  const float* Wg2   = (const float*)d_in[12];
  const float* bg2   = (const float*)d_in[13];
  const float* Wd1   = (const float*)d_in[14];
  const float* bd1   = (const float*)d_in[15];
  const float* Wd2   = (const float*)d_in[16];
  const float* bd2   = (const float*)d_in[17];
  const float* Wr1   = (const float*)d_in[18];
  const float* br1   = (const float*)d_in[19];
  const float* Wr2   = (const float*)d_in[20];
  const float* br2   = (const float*)d_in[21];
  float* etot = (float*)d_out;

  char* base = (char*)d_ws;
  size_t off = 0;
  auto take = [&](size_t bytes) -> char* {
    char* p = base + off;
    off = (off + bytes + 255) & ~(size_t)255;
    return p;
  };
  float*  h0    = (float*)take((size_t)NN * HH * 4);
  float*  h1    = (float*)take((size_t)NN * 3 * HH * 4);
  float*  agg   = (float*)take((size_t)NN * HH * 4);
  float*  dh1   = (float*)take((size_t)NN * 3 * HH * 4);
  bf16_t* WmsgT = (bf16_t*)take((size_t)LL * HH * HH * 2);
  bf16_t* WrbfT = (bf16_t*)take((size_t)LL * HH * RR * 2);
  bf16_t* Wg1T  = (bf16_t*)take((size_t)LL * HH * GIN * 2);
  bf16_t* WgvT  = (bf16_t*)take((size_t)LL * HH * HH * 2);
  bf16_t* Wg2T  = (bf16_t*)take((size_t)LL * HH * HH * 2);
  bf16_t* Wd1T  = (bf16_t*)take((size_t)LL * HH * HH * 2);
  bf16_t* Wd2T  = (bf16_t*)take((size_t)LL * HH * HH * 2);
  bf16_t* Wr1T  = (bf16_t*)take((size_t)LL * HH * HH * 2);

  const int gHH  = (HH * HH + 255) / 256;
  const int gRBF = (RR * HH + 255) / 256;
  const int gG1  = (GIN * HH + 255) / 256;
  for (int l = 0; l < LL; ++l) {
    k_convT<<<gHH, 256, 0, stream>>>(W_msg + l * HH * HH, WmsgT + l * HH * HH, HH, HH);
    k_convT<<<gRBF, 256, 0, stream>>>(W_rbf + l * RR * HH, WrbfT + l * HH * RR, RR, HH);
    k_convT<<<gG1, 256, 0, stream>>>(Wg1 + l * GIN * HH, Wg1T + l * HH * GIN, GIN, HH);
    k_convT<<<gHH, 256, 0, stream>>>(Wgv + l * HH * HH, WgvT + l * HH * HH, HH, HH);
    k_convT<<<gHH, 256, 0, stream>>>(Wg2 + l * HH * HH, Wg2T + l * HH * HH, HH, HH);
    k_convT<<<gHH, 256, 0, stream>>>(Wd1 + l * HH * HH, Wd1T + l * HH * HH, HH, HH);
    k_convT<<<gHH, 256, 0, stream>>>(Wd2 + l * HH * HH, Wd2T + l * HH * HH, HH, HH);
    k_convT<<<gHH, 256, 0, stream>>>(Wr1 + l * HH * HH, Wr1T + l * HH * HH, HH, HH);
  }

  k_zero<<<(out_size + 255) / 256, 256, 0, stream>>>(etot, out_size);
  k_init_h0<<<(NN * HH + 255) / 256, 256, 0, stream>>>(z, emb, h0);
  k_zero<<<(NN * 3 * HH + 255) / 256, 256, 0, stream>>>(h1, NN * 3 * HH);

  for (int l = 0; l < LL; ++l) {
    k_zero<<<(NN * HH + 255) / 256, 256, 0, stream>>>(agg, NN * HH);
    k_zero<<<(NN * 3 * HH + 255) / 256, 256, 0, stream>>>(dh1, NN * 3 * HH);
    k_edge<<<EE / TE, 256, 0, stream>>>(
        ei, pos, h0, h1,
        WmsgT + l * HH * HH, b_msg + l * HH,
        WrbfT + l * HH * RR,
        Wg1T + l * HH * GIN, bg1 + l * HH,
        WgvT + l * HH * HH,
        Wg2T + l * HH * HH, bg2 + l * HH,
        agg, dh1);
    k_node<<<NN / 16, 256, 0, stream>>>(
        agg, dh1,
        Wd1T + l * HH * HH, bd1 + l * HH,
        Wd2T + l * HH * HH, bd2 + l * HH,
        Wr1T + l * HH * HH, br1 + l * HH,
        Wr2 + l * HH, br2 + l,
        batch, h0, h1, etot);
  }
  k_ref<<<(NN + 255) / 256, 256, 0, stream>>>(z, batch, aref, etot);
}
